// CTS_CRFMF_1090921693477
// MI455X (gfx1250) — matrix operand order repaired and checked by reading, not compiled, not run on hardware
//
#include <hip/hip_runtime.h>
#include <hip/hip_bf16.h>
#include <math.h>

// ---------------- model constants ----------------
#define BN   4      // batch
#define IMS  224    // image size
#define NTOK 784    // 28*28 tokens
#define NCLS 9      // classes

typedef __bf16 bf16_t;
typedef bf16_t v16bf __attribute__((ext_vector_type(16)));
typedef float  v8f   __attribute__((ext_vector_type(8)));

// =====================================================================
// Generic batched WMMA GEMM (bf16 operands, f32 accumulate)
//   C[z][M,N] = scale * A[z][M,K] * B[z][K,N] (+bias[n]) (relu?) (+resid)
// B element (k,n) lives at Bb + k*sBk + n*sBn.
// z decomposition: i2 = z%nInner; i1 = (z/nInner)%nMid; i0 = z/(nInner*nMid).
// One wave computes a 16x64 C strip (4 WMMA accumulators share the A fragment).
// Requires: M%16==0, N%16==0, K%16==0 (true for every GEMM in this net).
// Main K-loop is branch-free with contiguous float4 loads; the K%32==16 tail
// (K=784 cases) is one masked step using clamped addresses + value selects,
// so EXEC never diverges around the WMMAs.
// Fragment layouts used here (wave32; A/B k order is contiguous, not ISA table order):
//   A (16x32 bf16): lane m=lane&15; e   -> k = 16*half+e
//   B (32x16 bf16): lane n=lane&15; e   -> k = 16*half+e
//   C (16x16 f32) : vgpr r -> row = r + 8*half, col = lane&15
// =====================================================================
template <bool BCONTIG>   // BCONTIG: sBk == 1 (linear layers, W rows contiguous)
__global__ __launch_bounds__(128) void wmma_gemm_k(
    const float* __restrict__ A, const float* __restrict__ Bm,
    float* __restrict__ C, const float* __restrict__ bias,
    const float* __restrict__ resid,
    int M, int N, int K,
    int lda, int sBn, int sBk, int ldc,
    long sA0, long sA1, long sA2,
    long sB0, long sB1, long sB2,
    long sC0, long sC1, long sC2,
    int nMid, int nInner, float scale, int relu)
{
  int z  = blockIdx.z;
  int i2 = z % nInner; int tz = z / nInner;
  int i1 = tz % nMid;  int i0 = tz / nMid;
  const float* Ab = A  + i0*sA0 + i1*sA1 + i2*sA2;
  const float* Bb = Bm + i0*sB0 + i1*sB1 + i2*sB2;
  float*       Cb = C  + i0*sC0 + i1*sC1 + i2*sC2;
  const float* Rb = resid ? (resid + i0*sC0 + i1*sC1 + i2*sC2) : (const float*)0;

  int wid  = threadIdx.x >> 5;
  int lane = threadIdx.x & 31;
  int half = lane >> 4;
  int lm   = lane & 15;
  int m0   = (blockIdx.y * 4 + wid) * 16;
  if (m0 >= M) return;                       // wave-uniform guard
  int nb   = blockIdx.x * 64;

  // clamped B column index per sub-tile (address-safe; junk columns never stored)
  int ncolS[4];
#pragma unroll
  for (int j = 0; j < 4; ++j) {
    int nc = nb + j * 16 + lm;
    ncolS[j] = (nc < N) ? nc : (N - 1);
  }

  v8f zero;
#pragma unroll
  for (int r = 0; r < 8; ++r) zero[r] = 0.f;
  v8f acc[4] = {zero, zero, zero, zero};

  const float* Arow = Ab + (size_t)(m0 + lm) * lda;
  int Kmain = K & ~31;

  // ------------------ unconditional main loop ------------------
  for (int kk = 0; kk < Kmain; kk += 32) {
    if (kk + 64 < K) __builtin_prefetch(Arow + kk + 64, 0, 1);  // global_prefetch_b8
    int b0 = kk + 16 * half;
    float4 a0 = *(const float4*)(Arow + b0);
    float4 a1 = *(const float4*)(Arow + b0 + 4);
    float4 a2 = *(const float4*)(Arow + b0 + 8);
    float4 a3 = *(const float4*)(Arow + b0 + 12);
    v16bf af;
    af[0]=(bf16_t)a0.x;  af[1]=(bf16_t)a0.y;  af[2]=(bf16_t)a0.z;  af[3]=(bf16_t)a0.w;
    af[4]=(bf16_t)a1.x;  af[5]=(bf16_t)a1.y;  af[6]=(bf16_t)a1.z;  af[7]=(bf16_t)a1.w;
    af[8]=(bf16_t)a2.x;  af[9]=(bf16_t)a2.y;  af[10]=(bf16_t)a2.z; af[11]=(bf16_t)a2.w;
    af[12]=(bf16_t)a3.x; af[13]=(bf16_t)a3.y; af[14]=(bf16_t)a3.z; af[15]=(bf16_t)a3.w;
    int kb = kk + 16 * half;
#pragma unroll
    for (int j = 0; j < 4; ++j) {
      v16bf bf;
      if (BCONTIG) {
        const float* Bp = Bb + (size_t)ncolS[j] * sBn + kb;
        float4 q0 = *(const float4*)(Bp);
        float4 q1 = *(const float4*)(Bp + 4);
        float4 q2 = *(const float4*)(Bp + 8);
        float4 q3 = *(const float4*)(Bp + 12);
        bf[0]=(bf16_t)q0.x;  bf[1]=(bf16_t)q0.y;  bf[2]=(bf16_t)q0.z;  bf[3]=(bf16_t)q0.w;
        bf[4]=(bf16_t)q1.x;  bf[5]=(bf16_t)q1.y;  bf[6]=(bf16_t)q1.z;  bf[7]=(bf16_t)q1.w;
        bf[8]=(bf16_t)q2.x;  bf[9]=(bf16_t)q2.y;  bf[10]=(bf16_t)q2.z; bf[11]=(bf16_t)q2.w;
        bf[12]=(bf16_t)q3.x; bf[13]=(bf16_t)q3.y; bf[14]=(bf16_t)q3.z; bf[15]=(bf16_t)q3.w;
      } else {
        const float* Bp = Bb + (size_t)ncolS[j] * sBn + (size_t)kb * sBk;
#pragma unroll
        for (int e = 0; e < 16; ++e) bf[e] = (bf16_t)Bp[(size_t)e * sBk];
      }
      acc[j] = __builtin_amdgcn_wmma_f32_16x16x32_bf16(
          false, af, false, bf, (short)0, acc[j], false, false);
    }
  }

  // ------------------ K%32==16 tail: one masked step ------------------
  if (Kmain < K) {
    int kk = Kmain;
    // A: half==0 lanes load k = kk..kk+15; half==1 lanes are zero (clamped addr + select)
    int b0 = kk;
    float amask = (half == 0) ? 1.f : 0.f;
    float4 a0 = *(const float4*)(Arow + b0);
    float4 a1 = *(const float4*)(Arow + b0 + 4);
    float4 a2 = *(const float4*)(Arow + b0 + 8);
    float4 a3 = *(const float4*)(Arow + b0 + 12);
    v16bf af;
    af[0]=(bf16_t)(a0.x*amask); af[1]=(bf16_t)(a0.y*amask); af[2]=(bf16_t)(a0.z*amask); af[3]=(bf16_t)(a0.w*amask);
    af[4]=(bf16_t)(a1.x*amask); af[5]=(bf16_t)(a1.y*amask); af[6]=(bf16_t)(a1.z*amask); af[7]=(bf16_t)(a1.w*amask);
    af[8]=(bf16_t)(a2.x*amask); af[9]=(bf16_t)(a2.y*amask); af[10]=(bf16_t)(a2.z*amask); af[11]=(bf16_t)(a2.w*amask);
    af[12]=(bf16_t)(a3.x*amask); af[13]=(bf16_t)(a3.y*amask); af[14]=(bf16_t)(a3.z*amask); af[15]=(bf16_t)(a3.w*amask);
    // B: half==0 lanes load k = kk..kk+15; half==1 lanes are zero (clamped addr + select)
    float bmask = (half == 0) ? 1.f : 0.f;
    int kb = kk;                                     // clamped (valid for both halves)
#pragma unroll
    for (int j = 0; j < 4; ++j) {
      v16bf bf;
      if (BCONTIG) {
        const float* Bp = Bb + (size_t)ncolS[j] * sBn + kb;
        float4 q0 = *(const float4*)(Bp);
        float4 q1 = *(const float4*)(Bp + 4);
        float4 q2 = *(const float4*)(Bp + 8);
        float4 q3 = *(const float4*)(Bp + 12);
        bf[0]=(bf16_t)(q0.x*bmask);  bf[1]=(bf16_t)(q0.y*bmask);
        bf[2]=(bf16_t)(q0.z*bmask);  bf[3]=(bf16_t)(q0.w*bmask);
        bf[4]=(bf16_t)(q1.x*bmask);  bf[5]=(bf16_t)(q1.y*bmask);
        bf[6]=(bf16_t)(q1.z*bmask);  bf[7]=(bf16_t)(q1.w*bmask);
        bf[8]=(bf16_t)(q2.x*bmask);  bf[9]=(bf16_t)(q2.y*bmask);
        bf[10]=(bf16_t)(q2.z*bmask); bf[11]=(bf16_t)(q2.w*bmask);
        bf[12]=(bf16_t)(q3.x*bmask); bf[13]=(bf16_t)(q3.y*bmask);
        bf[14]=(bf16_t)(q3.z*bmask); bf[15]=(bf16_t)(q3.w*bmask);
      } else {
        const float* Bp = Bb + (size_t)ncolS[j] * sBn + (size_t)kb * sBk;
#pragma unroll
        for (int e = 0; e < 16; ++e) bf[e] = (bf16_t)(Bp[(size_t)e * sBk] * bmask);
      }
      acc[j] = __builtin_amdgcn_wmma_f32_16x16x32_bf16(
          false, af, false, bf, (short)0, acc[j], false, false);
    }
  }

  // ------------------ fused epilogue: scale, bias, relu, residual ------------------
#pragma unroll
  for (int j = 0; j < 4; ++j) {
    if (nb + j * 16 < N) {                 // wave-uniform store guard (N%16==0)
      int ncol = nb + j * 16 + lm;
      float bv = bias ? bias[ncol] : 0.f;
#pragma unroll
      for (int r = 0; r < 8; ++r) {
        int mr  = m0 + r + 8 * half;
        float v = acc[j][r] * scale + bv;
        if (relu) v = fmaxf(v, 0.f);
        size_t o = (size_t)mr * ldc + ncol;
        if (Rb) v += Rb[o];
        Cb[o] = v;
      }
    }
  }
}

// =====================================================================
// Support kernels (VALU; negligible FLOPs)
// =====================================================================
__global__ void conv3x3_k(const float* __restrict__ in, float* __restrict__ out,
                          const float* __restrict__ w, const float* __restrict__ bias,
                          const float* __restrict__ g, const float* __restrict__ be,
                          const float* __restrict__ resid,
                          int Cin, int Cout, int H, int W, int relu, int total)
{
  int idx = blockIdx.x * blockDim.x + threadIdx.x;
  if (idx >= total) return;
  int x = idx % W; int t = idx / W;
  int y = t % H;   t /= H;
  int co = t % Cout; int b = t / Cout;
  float acc = bias[co];
  const float* wp = w + (size_t)co * Cin * 9;
  for (int ci = 0; ci < Cin; ++ci) {
    const float* ip = in + (size_t)(b * Cin + ci) * H * W;
    const float* wc = wp + ci * 9;
#pragma unroll
    for (int ky = 0; ky < 3; ++ky) {
      int iy = y + ky - 1;
      if ((unsigned)iy < (unsigned)H) {
#pragma unroll
        for (int kx = 0; kx < 3; ++kx) {
          int ix = x + kx - 1;
          if ((unsigned)ix < (unsigned)W)
            acc += ip[(size_t)iy * W + ix] * wc[ky * 3 + kx];
        }
      }
    }
  }
  float v = acc * g[co] + be[co];
  if (relu) v = fmaxf(v, 0.f);
  if (resid) v += resid[idx];
  out[idx] = v;
}

__global__ void maxpool2_k(const float* __restrict__ in, float* __restrict__ out,
                           int C, int Ho, int Wo, int total)
{
  int idx = blockIdx.x * blockDim.x + threadIdx.x;
  if (idx >= total) return;
  int x = idx % Wo; int t = idx / Wo;
  int y = t % Ho;   t /= Ho;
  int c = t % C;    int b = t / C;
  int Wi = Wo * 2;
  const float* p = in + ((size_t)(b * C + c) * (Ho * 2) + 2 * y) * Wi + 2 * x;
  out[idx] = fmaxf(fmaxf(p[0], p[1]), fmaxf(p[Wi], p[Wi + 1]));
}

// _pack: linear over channels then (b,c,(n1 p1),(n2 p2)) -> (b, n1*28+n2, (c2*p1+i)*p1+j)
__global__ void pack_k(const float* __restrict__ feat, const float* __restrict__ w,
                       const float* __restrict__ bias, float* __restrict__ out,
                       int Cin, int C2, int p1, int total)
{
  int idx = blockIdx.x * blockDim.x + threadIdx.x;
  if (idx >= total) return;
  int pd = C2 * p1 * p1;
  int cc = idx % pd; int t = idx / pd;
  int tok = t % NTOK; int b = t / NTOK;
  int c2 = cc / (p1 * p1); int r = cc % (p1 * p1);
  int i = r / p1, j = r % p1;
  int n1 = tok / 28, n2 = tok % 28;
  int H = p1 * 28;
  int row = n1 * p1 + i, col = n2 * p1 + j;
  float acc = bias[c2];
  for (int c = 0; c < Cin; ++c)
    acc += feat[((size_t)(b * Cin + c) * H + row) * H + col] * w[c2 * Cin + c];
  out[idx] = acc;
}

__global__ void tok_init_k(const float* __restrict__ h3, const float* __restrict__ pos,
                           float* __restrict__ t, int total)  // [4,784,256]
{
  int idx = blockIdx.x * blockDim.x + threadIdx.x;
  if (idx >= total) return;
  int c = idx % 256; int q = idx / 256;
  int n = q % NTOK;  int b = q / NTOK;
  t[idx] = h3[((size_t)b * 256 + c) * NTOK + n] + pos[n * 256 + c];
}

__global__ void layernorm_k(const float* __restrict__ in, float* __restrict__ out,
                            const float* __restrict__ g, const float* __restrict__ b, int d)
{
  __shared__ float red[256];
  __shared__ float s_mean, s_rstd;
  int row = blockIdx.x, tid = threadIdx.x;
  const float* x = in + (size_t)row * d;
  float* y = out + (size_t)row * d;
  float s = 0.f, sq = 0.f;
  for (int i = tid; i < d; i += 256) { float v = x[i]; s += v; sq += v * v; }
  red[tid] = s; __syncthreads();
  for (int o = 128; o > 0; o >>= 1) { if (tid < o) red[tid] += red[tid + o]; __syncthreads(); }
  float mean = red[0] / d; __syncthreads();
  red[tid] = sq; __syncthreads();
  for (int o = 128; o > 0; o >>= 1) { if (tid < o) red[tid] += red[tid + o]; __syncthreads(); }
  if (tid == 0) { s_mean = mean; s_rstd = rsqrtf(red[0] / d - mean * mean + 1e-5f); }
  __syncthreads();
  float m = s_mean, r = s_rstd;
  for (int i = tid; i < d; i += 256) y[i] = (x[i] - m) * r * g[i] + b[i];
}

__global__ void softmax_rows_k(float* __restrict__ x, int len)
{
  __shared__ float red[256];
  __shared__ float s_inv;
  int tid = threadIdx.x;
  float* p = x + (size_t)blockIdx.x * len;
  float m = -1e30f;
  for (int i = tid; i < len; i += 256) m = fmaxf(m, p[i]);
  red[tid] = m; __syncthreads();
  for (int o = 128; o > 0; o >>= 1) { if (tid < o) red[tid] = fmaxf(red[tid], red[tid + o]); __syncthreads(); }
  m = red[0]; __syncthreads();
  float s = 0.f;
  for (int i = tid; i < len; i += 256) { float e = __expf(p[i] - m); p[i] = e; s += e; }
  red[tid] = s; __syncthreads();
  for (int o = 128; o > 0; o >>= 1) { if (tid < o) red[tid] += red[tid + o]; __syncthreads(); }
  if (tid == 0) s_inv = 1.f / red[0];
  __syncthreads();
  float inv = s_inv;
  for (int i = tid; i < len; i += 256) p[i] *= inv;
}

// tokens(512) -> pixel map(8ch), 1x1 seg head (8->9), store unary + Q=softmax(unary)
__global__ void seg_unary_k(const float* __restrict__ t, const float* __restrict__ sw,
                            const float* __restrict__ sb, float* __restrict__ unary,
                            float* __restrict__ Q, int total)
{
  int idx = blockIdx.x * blockDim.x + threadIdx.x;
  if (idx >= total) return;
  int ww = idx % IMS; int tq = idx / IMS;
  int hh = tq % IMS;  int b  = tq / IMS;
  int n1 = hh >> 3, pi = hh & 7, n2 = ww >> 3, pj = ww & 7;
  int n = n1 * 28 + n2, pp = pi * 8 + pj;
  const float* tp = t + ((size_t)b * NTOK + n) * 512;
  float yv[8];
#pragma unroll
  for (int c = 0; c < 8; ++c) yv[c] = tp[c * 64 + pp];
  float lg[NCLS]; float mx = -1e30f;
  for (int o = 0; o < NCLS; ++o) {
    float u = sb[o];
#pragma unroll
    for (int c = 0; c < 8; ++c) u += sw[o * 8 + c] * yv[c];
    unary[((size_t)(b * NCLS + o) * IMS + hh) * IMS + ww] = u;
    lg[o] = u; mx = fmaxf(mx, u);
  }
  float s = 0.f;
  for (int o = 0; o < NCLS; ++o) { lg[o] = __expf(lg[o] - mx); s += lg[o]; }
  float inv = 1.f / s;
  for (int o = 0; o < NCLS; ++o)
    Q[((size_t)(b * NCLS + o) * IMS + hh) * IMS + ww] = lg[o] * inv;
}

// Q[b,c,224,224] -> Qp[b,c,h,n,16]  (pp = pi*8+pj, h = pp/16, dd = pp%16)
__global__ void qp_gather_k(const float* __restrict__ Q, float* __restrict__ Qp, int total)
{
  int idx = blockIdx.x * blockDim.x + threadIdx.x;
  if (idx >= total) return;
  int dd = idx & 15; int t = idx >> 4;
  int n = t % NTOK; t /= NTOK;
  int h = t & 3; t >>= 2;
  int c = t % NCLS; int b = t / NCLS;
  int pp = h * 16 + dd, pi = pp >> 3, pj = pp & 7;
  int n1 = n / 28, n2 = n % 28;
  Qp[idx] = Q[((size_t)(b * NCLS + c) * IMS + n1 * 8 + pi) * IMS + n2 * 8 + pj];
}

// per-row 16 -> relu(32) -> 16 mini-MLP (mf_ff), weights cached in LDS
__global__ void mf_ff_k(const float* __restrict__ in, float* __restrict__ out,
                        const float* __restrict__ w1, const float* __restrict__ b1,
                        const float* __restrict__ w2, const float* __restrict__ b2,
                        int rows)
{
  __shared__ float sw1[512], sw2[512], sb1[32], sb2[16];
  int tid = threadIdx.x;
  for (int i = tid; i < 512; i += blockDim.x) { sw1[i] = w1[i]; sw2[i] = w2[i]; }
  if (tid < 32) sb1[tid] = b1[tid];
  if (tid < 16) sb2[tid] = b2[tid];
  __syncthreads();
  int row = blockIdx.x * blockDim.x + tid;
  if (row >= rows) return;
  const float* x = in + (size_t)row * 16;
  float xv[16];
#pragma unroll
  for (int i = 0; i < 16; ++i) xv[i] = x[i];
  float hid[32];
  for (int j = 0; j < 32; ++j) {
    float a = sb1[j];
#pragma unroll
    for (int i = 0; i < 16; ++i) a += sw1[j * 16 + i] * xv[i];
    hid[j] = fmaxf(a, 0.f);
  }
  float* o = out + (size_t)row * 16;
  for (int k = 0; k < 16; ++k) {
    float a = sb2[k];
#pragma unroll
    for (int j = 0; j < 32; ++j) a += sw2[k * 32 + j] * hid[j];
    o[k] = a;
  }
}

// compat (9x9 over classes) + Q = softmax(unary - compat)
__global__ void compat_softmax_k(const float* __restrict__ ff, const float* __restrict__ unary,
                                 const float* __restrict__ cw, const float* __restrict__ cb,
                                 float* __restrict__ Q, int total)
{
  int idx = blockIdx.x * blockDim.x + threadIdx.x;
  if (idx >= total) return;
  int ww = idx % IMS; int t = idx / IMS;
  int hh = t % IMS;   int b = t / IMS;
  int n1 = hh >> 3, pi = hh & 7, n2 = ww >> 3, pj = ww & 7;
  int pp = pi * 8 + pj, h = pp >> 4, dd = pp & 15;
  int n = n1 * 28 + n2;
  float f[NCLS];
  for (int c = 0; c < NCLS; ++c)
    f[c] = ff[((((size_t)b * NCLS + c) * 4 + h) * NTOK + n) * 16 + dd];
  float lg[NCLS]; float mx = -1e30f;
  for (int o = 0; o < NCLS; ++o) {
    float cp = cb[o];
    for (int c = 0; c < NCLS; ++c) cp += cw[o * NCLS + c] * f[c];
    float u = unary[((size_t)(b * NCLS + o) * IMS + hh) * IMS + ww] - cp;
    lg[o] = u; mx = fmaxf(mx, u);
  }
  float s = 0.f;
  for (int o = 0; o < NCLS; ++o) { lg[o] = __expf(lg[o] - mx); s += lg[o]; }
  float inv = 1.f / s;
  for (int o = 0; o < NCLS; ++o)
    Q[((size_t)(b * NCLS + o) * IMS + hh) * IMS + ww] = lg[o] * inv;
}

// =====================================================================
// Host side
// =====================================================================
struct LinP  { const float *w, *b; };
struct LNP   { const float *g, *b; };
struct ConvP { const float *w, *b, *g, *be; };
struct EncP  { LNP n1, n2; LinP q, k, v, o, l1, l2; };

static void gemm_full(hipStream_t s, const float* A, const float* B, float* C,
                      const float* bias, const float* resid,
                      int M, int N, int K, int lda, int sBn, int sBk, int ldc,
                      long sA0, long sA1, long sA2,
                      long sB0, long sB1, long sB2,
                      long sC0, long sC1, long sC2,
                      int nMid, int nInner, int Z, float scale, int relu)
{
  dim3 g((N + 63) / 64, (M + 63) / 64, Z), b(128, 1, 1);
  if (sBk == 1)
    wmma_gemm_k<true><<<g, b, 0, s>>>(A, B, C, bias, resid, M, N, K, lda, sBn, sBk, ldc,
                                      sA0, sA1, sA2, sB0, sB1, sB2, sC0, sC1, sC2,
                                      nMid, nInner, scale, relu);
  else
    wmma_gemm_k<false><<<g, b, 0, s>>>(A, B, C, bias, resid, M, N, K, lda, sBn, sBk, ldc,
                                       sA0, sA1, sA2, sB0, sB1, sB2, sC0, sC1, sC2,
                                       nMid, nInner, scale, relu);
}

static void gemm_lin(hipStream_t s, const float* X, const float* W, const float* bias,
                     const float* resid, float* Y, int M, int N, int K, int relu)
{
  gemm_full(s, X, W, Y, bias, resid, M, N, K, K, K, 1, N,
            0, 0, 0, 0, 0, 0, 0, 0, 0, 1, 1, 1, 1.f, relu);
}

extern "C" void kernel_launch(void* const* d_in, const int* in_sizes, int n_in,
                              void* d_out, int out_size, void* d_ws, size_t ws_size,
                              hipStream_t stream)
{
  // ---------- parse inputs in setup_inputs() leaf order ----------
  int ii = 0;
  auto nx = [&]() { return (const float*)d_in[ii++]; };
  const float* x = nx();
  ConvP conv[4][3];                                   // c1, c2, cs per block
  for (int bl = 0; bl < 4; ++bl)
    for (int c = 0; c < 3; ++c) { conv[bl][c].w = nx(); conv[bl][c].b = nx();
                                  conv[bl][c].g = nx(); conv[bl][c].be = nx(); }
  LinP lin[3];
  for (int i = 0; i < 3; ++i) { lin[i].w = nx(); lin[i].b = nx(); }
  const float* pos = nx();
  auto rdEnc = [&](EncP& e) {
    e.n1.g = nx(); e.n1.b = nx(); e.n2.g = nx(); e.n2.b = nx();
    e.q.w = nx(); e.q.b = nx(); e.k.w = nx(); e.k.b = nx();
    e.v.w = nx(); e.v.b = nx(); e.o.w = nx(); e.o.b = nx();
    e.l1.w = nx(); e.l1.b = nx(); e.l2.w = nx(); e.l2.b = nx();
  };
  EncP trans0[2]; rdEnc(trans0[0]); rdEnc(trans0[1]);
  LinP stage_lin[3];
  for (int i = 0; i < 3; ++i) { stage_lin[i].w = nx(); stage_lin[i].b = nx(); }
  int snl[3] = {2, 2, 1};
  EncP sblk[3][2];
  for (int i = 0; i < 3; ++i) for (int l = 0; l < snl[i]; ++l) rdEnc(sblk[i][l]);
  EncP crf; rdEnc(crf);
  LinP seg;  seg.w = nx();  seg.b = nx();
  LNP  mfn;  mfn.g = nx();  mfn.b = nx();
  LinP mfq;  mfq.w = nx();  mfq.b = nx();
  LinP mfk;  mfk.w = nx();  mfk.b = nx();
  LinP comp; comp.w = nx(); comp.b = nx();
  LinP mfl1; mfl1.w = nx(); mfl1.b = nx();
  LinP mfl2; mfl2.w = nx(); mfl2.b = nx();
  (void)n_in; (void)in_sizes;

  // ---------- workspace layout (~420 MB; all f32) ----------
  char* base = (char*)d_ws; size_t off = 0;
  auto alloc = [&](size_t nf) { float* p = (float*)(base + off); off += nf * sizeof(float); return p; };
  float* convT1 = alloc(6422528);   float* convSk = alloc(6422528);
  float* h0 = alloc(6422528);       float* hp = alloc(1605632);
  float* h1 = alloc(3211264);       float* h2 = alloc(1605632);
  float* h3 = alloc(802816);
  float* cnn0 = alloc(1605632);     float* cnn1 = alloc(802816);
  float* cnn2 = alloc(401408);
  float* tokA = alloc(1605632);     float* tokB = alloc(1605632);
  float* tokX2 = alloc(1605632);    float* tokLN = alloc(1605632);
  float* tokQ = alloc(1605632);     float* tokK = alloc(1605632);
  float* tokV = alloc(1605632);     float* tokO = alloc(1605632);
  float* tokH = alloc(3211264);
  float* att  = alloc(39337984UL);  // transient attention, up to 16 heads
  float* attC = alloc(9834496UL);   // persistent CRF affinity (fits in 192MB L2)
  float* unary = alloc(1806336);    float* Qb = alloc(1806336);
  float* Qp = alloc(1806336);       float* msg = alloc(1806336);
  float* ffb = alloc(1806336);
  (void)ws_size;

  // ---------- CNN stem ----------
  auto runconv = [&](const float* in, float* out, ConvP* blk, int Cin, int Cout, int H, int W) {
    int tot = BN * Cout * H * W; int g = (tot + 255) / 256;
    conv3x3_k<<<g, 256, 0, stream>>>(in, convSk, blk[2].w, blk[2].b, blk[2].g, blk[2].be,
                                     nullptr, Cin, Cout, H, W, 0, tot);          // skip path
    conv3x3_k<<<g, 256, 0, stream>>>(in, convT1, blk[0].w, blk[0].b, blk[0].g, blk[0].be,
                                     nullptr, Cin, Cout, H, W, 1, tot);          // c1+relu
    conv3x3_k<<<g, 256, 0, stream>>>(convT1, out, blk[1].w, blk[1].b, blk[1].g, blk[1].be,
                                     convSk, Cout, Cout, H, W, 1, tot);          // c2+relu+skip
  };
  runconv(x, h0, conv[0], 3, 32, 224, 224);
  { int tot = BN * 32 * 112 * 112; maxpool2_k<<<(tot + 255) / 256, 256, 0, stream>>>(h0, hp, 32, 112, 112, tot); }
  runconv(hp, h1, conv[1], 32, 64, 112, 112);
  { int tot = BN * 64 * 56 * 56;  maxpool2_k<<<(tot + 255) / 256, 256, 0, stream>>>(h1, hp, 64, 56, 56, tot); }
  runconv(hp, h2, conv[2], 64, 128, 56, 56);
  { int tot = BN * 128 * 28 * 28; maxpool2_k<<<(tot + 255) / 256, 256, 0, stream>>>(h2, hp, 128, 28, 28, tot); }
  runconv(hp, h3, conv[3], 128, 256, 28, 28);
  { int tot = BN * NTOK * 512; pack_k<<<(tot + 255) / 256, 256, 0, stream>>>(h0, lin[0].w, lin[0].b, cnn0, 32, 8, 8, tot); }
  { int tot = BN * NTOK * 256; pack_k<<<(tot + 255) / 256, 256, 0, stream>>>(h1, lin[1].w, lin[1].b, cnn1, 64, 16, 4, tot); }
  { int tot = BN * NTOK * 128; pack_k<<<(tot + 255) / 256, 256, 0, stream>>>(h2, lin[2].w, lin[2].b, cnn2, 128, 32, 2, tot); }

  // ---------- tokens ----------
  { int tot = BN * NTOK * 256; tok_init_k<<<(tot + 255) / 256, 256, 0, stream>>>(h3, pos, tokA, tot); }

  const int M = BN * NTOK;
  auto runEnc = [&](float* xt, const EncP& e, int d, int heads) {
    int dh = d / heads;
    layernorm_k<<<M, 256, 0, stream>>>(xt, tokLN, e.n1.g, e.n1.b, d);
    gemm_lin(stream, tokLN, e.q.w, e.q.b, nullptr, tokQ, M, d, d, 0);
    gemm_lin(stream, tokLN, e.k.w, e.k.b, nullptr, tokK, M, d, d, 0);
    gemm_lin(stream, tokLN, e.v.w, e.v.b, nullptr, tokV, M, d, d, 0);
    // S[b,h] = (q k^T)/sqrt(dh)   (B = K matrix, contiguous along k -> BCONTIG)
    gemm_full(stream, tokQ, tokK, att, nullptr, nullptr, NTOK, NTOK, dh,
              d, d, 1, NTOK,
              (long)NTOK * d, 0, dh,
              (long)NTOK * d, 0, dh,
              (long)heads * NTOK * NTOK, 0, (long)NTOK * NTOK,
              1, heads, BN * heads, 1.0f / sqrtf((float)dh), 0);
    softmax_rows_k<<<BN * heads * NTOK, 256, 0, stream>>>(att, NTOK);
    // O[b,h] = S * V  (B strided: sBk = d)
    gemm_full(stream, att, tokV, tokO, nullptr, nullptr, NTOK, dh, NTOK,
              NTOK, 1, d, d,
              (long)heads * NTOK * NTOK, 0, (long)NTOK * NTOK,
              (long)NTOK * d, 0, dh,
              (long)NTOK * d, 0, dh,
              1, heads, BN * heads, 1.f, 0);
    gemm_lin(stream, tokO, e.o.w, e.o.b, xt, tokX2, M, d, d, 0);   // + residual x
    layernorm_k<<<M, 256, 0, stream>>>(tokX2, tokLN, e.n2.g, e.n2.b, d);
    gemm_lin(stream, tokLN, e.l1.w, e.l1.b, nullptr, tokH, M, 2 * d, d, 1);
    gemm_lin(stream, tokH, e.l2.w, e.l2.b, tokX2, xt, M, d, 2 * d, 0); // + residual
  };

  float* cur = tokA; float* oth = tokB;
  for (int l = 0; l < 2; ++l) runEnc(cur, trans0[l], 256, 8);
  int sdim[3] = {128, 256, 512}, sheads[3] = {4, 8, 16};
  float* cnnadd[3] = {cnn2, cnn1, cnn0};     // stage i adds cnn_out[2-i]
  int prevd = 256;
  for (int i = 0; i < 3; ++i) {
    gemm_lin(stream, cur, stage_lin[i].w, stage_lin[i].b, cnnadd[i], oth, M, sdim[i], prevd, 0);
    float* tswap = cur; cur = oth; oth = tswap;
    for (int l = 0; l < snl[i]; ++l) runEnc(cur, sblk[i][l], sdim[i], sheads[i]);
    prevd = sdim[i];
  }
  runEnc(cur, crf, 512, 4);   // EncoderLayer_crf

  // ---------- seg head: unary + initial Q ----------
  int totpx = BN * IMS * IMS;
  seg_unary_k<<<(totpx + 255) / 256, 256, 0, stream>>>(cur, seg.w, seg.b, unary, Qb, totpx);

  // ---------- CRF token-affinity (4 heads, dh=128) ----------
  layernorm_k<<<M, 256, 0, stream>>>(cur, tokLN, mfn.g, mfn.b, 512);
  gemm_lin(stream, tokLN, mfq.w, mfq.b, nullptr, tokQ, M, 512, 512, 0);
  gemm_lin(stream, tokLN, mfk.w, mfk.b, nullptr, tokK, M, 512, 512, 0);
  gemm_full(stream, tokQ, tokK, attC, nullptr, nullptr, NTOK, NTOK, 128,
            512, 512, 1, NTOK,
            (long)NTOK * 512, 0, 128,
            (long)NTOK * 512, 0, 128,
            (long)4 * NTOK * NTOK, 0, (long)NTOK * NTOK,
            1, 4, BN * 4, 1.0f / sqrtf(128.f), 0);
  softmax_rows_k<<<BN * 4 * NTOK, 256, 0, stream>>>(attC, NTOK);

  // ---------- 10 mean-field iterations (attC reread from L2 each iter) ----------
  int qtot = BN * NCLS * 4 * NTOK * 16;
  int rows = BN * NCLS * 4 * NTOK;
  for (int it = 0; it < 10; ++it) {
    qp_gather_k<<<(qtot + 255) / 256, 256, 0, stream>>>(Qb, Qp, qtot);
    // msg[b,c,h] = attC[b,h] (784x784) * Qp[b,c,h] (784x16)   (B strided: sBk = 16)
    gemm_full(stream, attC, Qp, msg, nullptr, nullptr, NTOK, 16, NTOK,
              NTOK, 1, 16, 16,
              (long)4 * NTOK * NTOK, 0, (long)NTOK * NTOK,
              (long)NCLS * 4 * NTOK * 16, (long)4 * NTOK * 16, (long)NTOK * 16,
              (long)NCLS * 4 * NTOK * 16, (long)4 * NTOK * 16, (long)NTOK * 16,
              NCLS, 4, BN * NCLS * 4, 1.f, 0);
    mf_ff_k<<<(rows + 255) / 256, 256, 0, stream>>>(msg, ffb, mfl1.w, mfl1.b, mfl2.w, mfl2.b, rows);
    compat_softmax_k<<<(totpx + 255) / 256, 256, 0, stream>>>(ffb, unary, comp.w, comp.b, Qb, totpx);
  }

  hipMemcpyAsync(d_out, Qb, (size_t)out_size * sizeof(float), hipMemcpyDeviceToDevice, stream);
}
